// MultiHeadAttention_57303453663372
// MI455X (gfx1250) — matrix operand order repaired and checked by reading, not compiled, not run on hardware
//
#include <hip/hip_runtime.h>
#include <hip/hip_bf16.h>

// ---------------------------------------------------------------------------
// MHA forward for MI455X (gfx1250, wave32, WMMA).
// D=1024, H=16, Dk=64, B=2, S=2048  ->  M = B*S = 4096 rows.
// All GEMMs in bf16 x bf16 -> f32 accumulate via v_wmma_f32_16x16x32_bf16.
// LDS staging uses GLOBAL_LOAD_ASYNC_TO_LDS_B128 (ASYNCcnt) when the
// toolchain exposes the builtin; falls back to load+ds_store otherwise.
// ---------------------------------------------------------------------------

typedef __bf16 bf16_t;
typedef __attribute__((ext_vector_type(16))) __bf16 v16bf;
typedef __attribute__((ext_vector_type(8)))  __bf16 v8bf;
typedef __attribute__((ext_vector_type(8)))  float  v8f;

#define D_MODEL 1024
#define NUM_HEADS 16
#define D_K 64
#define BATCH 2
#define SEQ 2048
#define MROWS (BATCH * SEQ)   // 4096

// ---- gfx1250 async-to-LDS path (guarded; fallback keeps compiling) --------
#if defined(__has_builtin)
#if __has_builtin(__builtin_amdgcn_global_load_async_to_lds_b128) && \
    __has_builtin(__builtin_amdgcn_s_wait_asynccnt)
#define HAVE_ASYNC_LDS 1
#endif
#endif
#ifndef HAVE_ASYNC_LDS
#define HAVE_ASYNC_LDS 0
#endif

#if HAVE_ASYNC_LDS
// Builtin signature (from hipcc diagnostic): param0 is
//   "__attribute__((vector_size(16))) int __device__ *"  (GCC vector, AS1)
typedef int b128v __attribute__((vector_size(16)));
typedef __attribute__((address_space(1))) b128v gb128v;
typedef __attribute__((address_space(3))) b128v sb128v;
static __device__ inline void async_copy16(const bf16_t* g, bf16_t* l) {
  __builtin_amdgcn_global_load_async_to_lds_b128((gb128v*)g, (sb128v*)l, 0, 0);
}
static __device__ inline void wait_async0() {
  __builtin_amdgcn_s_wait_asynccnt(0);
}
#else
static __device__ inline void wait_async0() {}
#endif

static __device__ inline v8f wmma_bf16(v16bf a, v16bf b, v8f c) {
  // (neg_a, A, neg_b, B, c_mod, C, reuse_a, reuse_b)
  return __builtin_amdgcn_wmma_f32_16x16x32_bf16(false, a, false, b, (short)0, c,
                                                 false, false);
}

static __device__ inline v16bf combine16(v8bf lo, v8bf hi) {
  v16bf r;
#pragma unroll
  for (int e = 0; e < 8; ++e) { r[e] = lo[e]; r[e + 8] = hi[e]; }
  return r;
}

static __device__ inline v8f vzero8() {
  v8f z = {0.f, 0.f, 0.f, 0.f, 0.f, 0.f, 0.f, 0.f};
  return z;
}

// ---------------------------------------------------------------------------
// fp32 -> bf16 cast
// ---------------------------------------------------------------------------
__global__ __launch_bounds__(256) void cast_f32_to_bf16(const float* __restrict__ in,
                                                        bf16_t* __restrict__ out,
                                                        int n) {
  int i = blockIdx.x * 256 + threadIdx.x;
  if (i < n) out[i] = (bf16_t)in[i];
}

// ---------------------------------------------------------------------------
// GEMM: C[M,N] = A[M,K] * W[N,K]^T   (A, W bf16; accumulate f32)
// Block = 256 threads = 8 waves; block tile 128x128; k-step 32, double-buffered.
// Waves arranged 4(M) x 2(N); each wave computes 32x64 = 2x4 wmma tiles.
// mode 0: write bf16 into [B*H][S][Dk] layout (QKV projections)
// mode 1: write f32 row-major [M][N] (final output)
// ---------------------------------------------------------------------------
__global__ __launch_bounds__(256) void gemm_bf16_wmma(const bf16_t* __restrict__ A,
                                                      const bf16_t* __restrict__ W,
                                                      float* __restrict__ outF,
                                                      bf16_t* __restrict__ outH,
                                                      int M, int N, int K, int mode) {
  __shared__ bf16_t As[2][128][40];  // 32 k-cols + 8 pad; row stride 80 B
  __shared__ bf16_t Bs[2][128][40];

  const int tid  = threadIdx.x;
  const int lane = tid & 31;
  const int wv   = tid >> 5;
  const int wr   = wv >> 1;        // 0..3  (M direction)
  const int wc   = wv & 1;         // 0..1  (N direction)
  const int half = lane >> 4;      // 0/1
  const int l15  = lane & 15;

  const int m0 = blockIdx.x * 128;
  const int n0 = blockIdx.y * 128;

  v8f acc[2][4];
#pragma unroll
  for (int i = 0; i < 2; ++i)
#pragma unroll
    for (int j = 0; j < 4; ++j) acc[i][j] = vzero8();

  const int srow = tid >> 1;            // 0..127 staging row
  const int sch  = (tid & 1) * 16;      // 0 / 16 staging col half

  const bf16_t* gA = &A[(size_t)(m0 + srow) * K + sch];
  const bf16_t* gW = &W[(size_t)(n0 + srow) * K + sch];

  auto stage = [&](int k0, int buf) {
#if HAVE_ASYNC_LDS
    async_copy16(gA + k0,     &As[buf][srow][sch]);
    async_copy16(gA + k0 + 8, &As[buf][srow][sch + 8]);
    async_copy16(gW + k0,     &Bs[buf][srow][sch]);
    async_copy16(gW + k0 + 8, &Bs[buf][srow][sch + 8]);
#else
    v8bf a0 = *reinterpret_cast<const v8bf*>(gA + k0);
    v8bf a1 = *reinterpret_cast<const v8bf*>(gA + k0 + 8);
    v8bf b0 = *reinterpret_cast<const v8bf*>(gW + k0);
    v8bf b1 = *reinterpret_cast<const v8bf*>(gW + k0 + 8);
    *reinterpret_cast<v8bf*>(&As[buf][srow][sch])     = a0;
    *reinterpret_cast<v8bf*>(&As[buf][srow][sch + 8]) = a1;
    *reinterpret_cast<v8bf*>(&Bs[buf][srow][sch])     = b0;
    *reinterpret_cast<v8bf*>(&Bs[buf][srow][sch + 8]) = b1;
#endif
    if (k0 + 32 < K) {  // prefetch tile after next into L2
      __builtin_prefetch(gA + k0 + 32, 0, 1);
      __builtin_prefetch(gW + k0 + 32, 0, 1);
    }
  };

  stage(0, 0);
  int buf = 0;
  for (int k0 = 0; k0 < K; k0 += 32, buf ^= 1) {
    wait_async0();       // my async transfers for `buf` have landed
    __syncthreads();     // everyone's landed; everyone done reading buf^1
    if (k0 + 32 < K) stage(k0 + 32, buf ^ 1);

    // ---- build fragments (ISA 7.12.2 layouts) ----
    v16bf af[2], bfr[4];
#pragma unroll
    for (int i = 0; i < 2; ++i) {
      const int row = wr * 32 + i * 16 + l15;
      v8bf lo = *reinterpret_cast<const v8bf*>(&As[buf][row][half * 8]);
      v8bf hi = *reinterpret_cast<const v8bf*>(&As[buf][row][16 + half * 8]);
      af[i] = combine16(lo, hi);
    }
#pragma unroll
    for (int j = 0; j < 4; ++j) {
      const int n = wc * 64 + j * 16 + l15;
      v8bf lo = *reinterpret_cast<const v8bf*>(&Bs[buf][n][half * 8]);
      v8bf hi = *reinterpret_cast<const v8bf*>(&Bs[buf][n][16 + half * 8]);
      bfr[j] = combine16(lo, hi);
    }

#pragma unroll
    for (int i = 0; i < 2; ++i)
#pragma unroll
      for (int j = 0; j < 4; ++j) acc[i][j] = wmma_bf16(af[i], bfr[j], acc[i][j]);
  }

  // ---- epilogue: base pointer per row-tile, constant-offset stores ----
  if (mode == 0) {
    const int b  = m0 >> 11;                // batch fixed per block (128 | 2048)
    const int hh = (n0 >> 6) + wc;          // head fixed per wave column
#pragma unroll
    for (int i = 0; i < 2; ++i) {
      const int sbase = (m0 & (SEQ - 1)) + wr * 32 + i * 16 + half * 8;
      bf16_t* p = outH + ((((size_t)b * NUM_HEADS + hh) * SEQ + sbase) << 6) + l15;
#pragma unroll
      for (int j = 0; j < 4; ++j)
#pragma unroll
        for (int r = 0; r < 8; ++r)
          p[(size_t)r * D_K + j * 16] = (bf16_t)acc[i][j][r];
    }
  } else {
#pragma unroll
    for (int i = 0; i < 2; ++i) {
      const int mrow = m0 + wr * 32 + i * 16 + half * 8;
      float* p = outF + (size_t)mrow * N + n0 + wc * 64 + l15;
#pragma unroll
      for (int j = 0; j < 4; ++j)
#pragma unroll
        for (int r = 0; r < 8; ++r)
          p[(size_t)r * N + j * 16] = acc[i][j][r];
    }
  }
}

// ---------------------------------------------------------------------------
// Flash attention: per block = one (b,h) pair x 64 query rows; 4 waves,
// each wave owns 16 query rows. Streams keys in chunks of 32; K/V chunk
// staged in LDS shared by the 4 waves. Online softmax in C-layout registers;
// P goes through per-wave LDS scratch to reach A-layout.
// ---------------------------------------------------------------------------
__global__ __launch_bounds__(128) void flash_attn_wmma(const bf16_t* __restrict__ Qb,
                                                       const bf16_t* __restrict__ Kb,
                                                       const bf16_t* __restrict__ Vb,
                                                       bf16_t* __restrict__ ctxb) {
  __shared__ bf16_t Ks[32][72];       // [key][dk], row stride 144 B
  __shared__ bf16_t Vts[64][40];      // [dk][key] transposed, row stride 80 B
  __shared__ bf16_t Ps[4][16][40];    // per-wave P scratch [qrow][key], 80 B rows

  const int tid  = threadIdx.x;
  const int lane = tid & 31;
  const int wv   = tid >> 5;
  const int half = lane >> 4;
  const int l15  = lane & 15;

  const int bh    = blockIdx.x;                 // 0..31  (b*16 + h)
  const int qbase = blockIdx.y * 64 + wv * 16;  // this wave's 16 query rows

  // ---- load Q fragments once (A-layout), fold in 1/sqrt(Dk)=0.125 (exact) ----
  const bf16_t* qrow = Qb + ((size_t)bh * SEQ + qbase + l15) * D_K;
  v16bf qa0 = combine16(*reinterpret_cast<const v8bf*>(qrow + half * 8),
                        *reinterpret_cast<const v8bf*>(qrow + 16 + half * 8));
  v16bf qa1 = combine16(*reinterpret_cast<const v8bf*>(qrow + 32 + half * 8),
                        *reinterpret_cast<const v8bf*>(qrow + 48 + half * 8));
#pragma unroll
  for (int e = 0; e < 16; ++e) {
    qa0[e] = (bf16_t)((float)qa0[e] * 0.125f);
    qa1[e] = (bf16_t)((float)qa1[e] * 0.125f);
  }

  v8f acc[4];
#pragma unroll
  for (int t = 0; t < 4; ++t) acc[t] = vzero8();
  float Mr[8], Lr[8];
#pragma unroll
  for (int r = 0; r < 8; ++r) { Mr[r] = -3.0e38f; Lr[r] = 0.f; }

  const int krow = tid >> 2;           // 0..31 key row for staging
  const int kcb  = (tid & 3) * 16;     // dk col base for staging
  const bf16_t* gK = Kb + ((size_t)bh * SEQ + krow) * D_K + kcb;
  const bf16_t* gV = Vb + ((size_t)bh * SEQ + krow) * D_K + kcb;

  for (int j = 0; j < SEQ; j += 32) {
    __syncthreads();  // previous chunk fully consumed
    // ---- stage K (async straight copy) and V (transposed, VALU path) ----
    {
      const size_t goff = (size_t)j * D_K;
#if HAVE_ASYNC_LDS
      async_copy16(gK + goff,     &Ks[krow][kcb]);
      async_copy16(gK + goff + 8, &Ks[krow][kcb + 8]);
#else
      v8bf k0 = *reinterpret_cast<const v8bf*>(gK + goff);
      v8bf k1 = *reinterpret_cast<const v8bf*>(gK + goff + 8);
      *reinterpret_cast<v8bf*>(&Ks[krow][kcb])     = k0;
      *reinterpret_cast<v8bf*>(&Ks[krow][kcb + 8]) = k1;
#endif
      v8bf v0 = *reinterpret_cast<const v8bf*>(gV + goff);
      v8bf v1 = *reinterpret_cast<const v8bf*>(gV + goff + 8);
#pragma unroll
      for (int e = 0; e < 8; ++e) {
        Vts[kcb + e][krow]     = v0[e];
        Vts[kcb + 8 + e][krow] = v1[e];
      }
    }
    wait_async0();
    __syncthreads();

    // ---- scores: S[st] = (Q*0.125) @ K^T  for keys j+st*16 .. +15 ----
    v8f sc[2];
#pragma unroll
    for (int st = 0; st < 2; ++st) {
      const int kr = st * 16 + l15;
      v16bf kb0 = combine16(*reinterpret_cast<const v8bf*>(&Ks[kr][half * 8]),
                            *reinterpret_cast<const v8bf*>(&Ks[kr][16 + half * 8]));
      v16bf kb1 = combine16(*reinterpret_cast<const v8bf*>(&Ks[kr][32 + half * 8]),
                            *reinterpret_cast<const v8bf*>(&Ks[kr][48 + half * 8]));
      v8f s = vzero8();
      s = wmma_bf16(qa0, kb0, s);
      s = wmma_bf16(qa1, kb1, s);
      sc[st] = s;
    }

    // ---- online softmax per row (row r / r+8 lives in 16-lane group) ----
#pragma unroll
    for (int r = 0; r < 8; ++r) {
      float v0 = sc[0][r], v1 = sc[1][r];
      float mx = fmaxf(v0, v1);
      mx = fmaxf(mx, __shfl_xor(mx, 1, 16));
      mx = fmaxf(mx, __shfl_xor(mx, 2, 16));
      mx = fmaxf(mx, __shfl_xor(mx, 4, 16));
      mx = fmaxf(mx, __shfl_xor(mx, 8, 16));
      const float nm   = fmaxf(Mr[r], mx);
      const float corr = __expf(Mr[r] - nm);
      Mr[r] = nm;
      const float p0 = __expf(v0 - nm);
      const float p1 = __expf(v1 - nm);
      float rs = p0 + p1;
      rs += __shfl_xor(rs, 1, 16);
      rs += __shfl_xor(rs, 2, 16);
      rs += __shfl_xor(rs, 4, 16);
      rs += __shfl_xor(rs, 8, 16);
      Lr[r] = Lr[r] * corr + rs;
#pragma unroll
      for (int t = 0; t < 4; ++t) acc[t][r] = acc[t][r] * corr;
      const int prow = r + half * 8;
      Ps[wv][prow][l15]      = (bf16_t)p0;
      Ps[wv][prow][16 + l15] = (bf16_t)p1;
    }
    asm volatile("" ::: "memory");  // keep P stores ordered before reloads

    // ---- ctx += P(16x32) @ V(32x64) ----
    v16bf pa = combine16(*reinterpret_cast<const v8bf*>(&Ps[wv][l15][half * 8]),
                         *reinterpret_cast<const v8bf*>(&Ps[wv][l15][16 + half * 8]));
#pragma unroll
    for (int t = 0; t < 4; ++t) {
      const int vr = t * 16 + l15;
      v16bf vbf = combine16(*reinterpret_cast<const v8bf*>(&Vts[vr][half * 8]),
                            *reinterpret_cast<const v8bf*>(&Vts[vr][16 + half * 8]));
      acc[t] = wmma_bf16(pa, vbf, acc[t]);
    }
  }

  // ---- normalize + write ctx as [b*S+s][h*64+dk] bf16 ----
  const int b = bh >> 4, h = bh & 15;
  bf16_t* p = ctxb + ((size_t)b * SEQ + qbase + half * 8) * D_MODEL + h * D_K + l15;
#pragma unroll
  for (int r = 0; r < 8; ++r) {
    const float inv = 1.0f / Lr[r];
#pragma unroll
    for (int t = 0; t < 4; ++t)
      p[(size_t)r * D_MODEL + t * 16] = (bf16_t)(acc[t][r] * inv);
  }
}

// ---------------------------------------------------------------------------
// Host-side launch
// ---------------------------------------------------------------------------
extern "C" void kernel_launch(void* const* d_in, const int* in_sizes, int n_in,
                              void* d_out, int out_size, void* d_ws, size_t ws_size,
                              hipStream_t stream) {
  const float* x  = (const float*)d_in[0];
  const float* Wq = (const float*)d_in[1];
  const float* Wk = (const float*)d_in[2];
  const float* Wv = (const float*)d_in[3];
  const float* Wo = (const float*)d_in[4];

  const int xN = MROWS * D_MODEL;        // 4096*1024
  const int wN = D_MODEL * D_MODEL;      // 1024*1024

  char* ws = (char*)d_ws;
  size_t off = 0;
  bf16_t* xb  = (bf16_t*)(ws + off); off += (size_t)xN * 2;
  bf16_t* wqb = (bf16_t*)(ws + off); off += (size_t)wN * 2;
  bf16_t* wkb = (bf16_t*)(ws + off); off += (size_t)wN * 2;
  bf16_t* wvb = (bf16_t*)(ws + off); off += (size_t)wN * 2;
  bf16_t* wob = (bf16_t*)(ws + off); off += (size_t)wN * 2;
  bf16_t* Qb  = (bf16_t*)(ws + off); off += (size_t)xN * 2;   // [B*H][S][Dk]
  bf16_t* Kb  = (bf16_t*)(ws + off); off += (size_t)xN * 2;
  bf16_t* Vb  = (bf16_t*)(ws + off); off += (size_t)xN * 2;
  bf16_t* ctxb = (bf16_t*)(ws + off); off += (size_t)xN * 2;  // [M][D]

  // 1) casts to bf16
  cast_f32_to_bf16<<<(xN + 255) / 256, 256, 0, stream>>>(x, xb, xN);
  cast_f32_to_bf16<<<(wN + 255) / 256, 256, 0, stream>>>(Wq, wqb, wN);
  cast_f32_to_bf16<<<(wN + 255) / 256, 256, 0, stream>>>(Wk, wkb, wN);
  cast_f32_to_bf16<<<(wN + 255) / 256, 256, 0, stream>>>(Wv, wvb, wN);
  cast_f32_to_bf16<<<(wN + 255) / 256, 256, 0, stream>>>(Wo, wob, wN);

  // 2) QKV projections (write head-split bf16)
  dim3 ggrid(MROWS / 128, D_MODEL / 128);  // 32 x 8
  gemm_bf16_wmma<<<ggrid, 256, 0, stream>>>(xb, wqb, nullptr, Qb, MROWS, D_MODEL, D_MODEL, 0);
  gemm_bf16_wmma<<<ggrid, 256, 0, stream>>>(xb, wkb, nullptr, Kb, MROWS, D_MODEL, D_MODEL, 0);
  gemm_bf16_wmma<<<ggrid, 256, 0, stream>>>(xb, wvb, nullptr, Vb, MROWS, D_MODEL, D_MODEL, 0);

  // 3) flash attention -> ctx bf16 [M][D]
  dim3 agrid(BATCH * NUM_HEADS, SEQ / 64);  // 32 x 32
  flash_attn_wmma<<<agrid, 128, 0, stream>>>(Qb, Kb, Vb, ctxb);

  // 4) output projection -> fp32 d_out
  gemm_bf16_wmma<<<ggrid, 256, 0, stream>>>(ctxb, wob, (float*)d_out, nullptr,
                                            MROWS, D_MODEL, D_MODEL, 1);
}
